// GCNModel_20040317403818
// MI455X (gfx1250) — compile-verified, matrix operand order checked by reading, not run on hardware
//
#include <hip/hip_runtime.h>
#include <hip/hip_bf16.h>
#include <stddef.h>

typedef __attribute__((ext_vector_type(16))) __bf16 v16bf;
typedef __attribute__((ext_vector_type(8)))  __bf16 v8bf;
typedef __attribute__((ext_vector_type(8)))  float  v8f;

#define N_NODES   100000
#define N_EDGES   1600000
#define N_FEAT    128
#define HIDDEN    128
#define N_CLASSES 10
#define N_GRAPHS  64
#define MTILES_PER_WAVE 4

// ---------- helpers ----------
__device__ __forceinline__ __bf16 to_bf16(float f) {
    union { float f; unsigned u; } x; x.f = f;
    unsigned r = x.u + 0x7FFFu + ((x.u >> 16) & 1u);   // round-to-nearest-even
    unsigned short h = (unsigned short)(r >> 16);
    __bf16 b;
    __builtin_memcpy(&b, &h, 2);
    return b;
}

// ---------- kernels ----------
__global__ void zero_f32(float* __restrict__ p, size_t n) {
    size_t i = (size_t)blockIdx.x * blockDim.x + threadIdx.x;
    if (i < n) p[i] = 0.0f;
}

__global__ void f32_to_bf16_k(const float* __restrict__ in, __bf16* __restrict__ out, size_t n) {
    size_t i = (size_t)blockIdx.x * blockDim.x + threadIdx.x;
    if (i < n) out[i] = to_bf16(in[i]);
}

// Wt[n*128 + k] = (bf16) W[k*128 + n]   (128x128 weight transpose + convert)
__global__ void prep_weight_k(const float* __restrict__ W, __bf16* __restrict__ Wt) {
    int i = blockIdx.x * blockDim.x + threadIdx.x;   // 16384 elements
    if (i < 128 * 128) {
        int n = i >> 7, k = i & 127;
        Wt[i] = to_bf16(W[k * 128 + n]);
    }
}

__global__ void degree_k(const int* __restrict__ dst, float* __restrict__ deg, int E) {
    int i = blockIdx.x * blockDim.x + threadIdx.x;
    if (i < E) atomicAdd(&deg[dst[i]], 1.0f);
}

__global__ void dis_k(const float* __restrict__ deg, float* __restrict__ dis, int N) {
    int i = blockIdx.x * blockDim.x + threadIdx.x;
    if (i < N) dis[i] = rsqrtf(deg[i] + 1.0f);       // degree includes self-loop
}

// C[M,128] (f32) = A[M,128] (bf16) @ W[128,128]  via Wt (bf16, transposed).
// block = 256 threads (8 waves). Each wave owns a 16-wide N-slice (n0 = wave*16),
// hoists all 4 B fragments (full K=128 of its W columns) into registers once,
// then sweeps MTILES_PER_WAVE row-tiles of 16 rows: 16 WMMAs per wave,
// weight loads amortized 4x. Tile guard is wave-uniform -> EXEC stays all-1s.
__global__ void gemm_wmma_bf16(const __bf16* __restrict__ A,
                               const __bf16* __restrict__ Wt,
                               float* __restrict__ C, int M) {
    const int wave  = threadIdx.x >> 5;
    const int lane  = threadIdx.x & 31;
    const int n0    = wave << 4;               // 8 waves cover N = 128
    const int m     = lane & 15;
    const int khalf = lane >> 4;               // 0: K {0..7,16..23}, 1: K {8..15,24..31}

    // --- load B fragments once (column n0+m of W, contiguous in k) ---
    const __bf16* brow = Wt + (size_t)(n0 + m) * 128;
    v16bf bfrag[4];
    #pragma unroll
    for (int s = 0; s < 4; ++s) {
        const int kk = s * 32;
        v8bf blo = *(const v8bf*)(brow + kk + khalf * 8);
        v8bf bhi = *(const v8bf*)(brow + kk + 16 + khalf * 8);
        #pragma unroll
        for (int j = 0; j < 8; ++j) { bfrag[s][j] = blo[j]; bfrag[s][8 + j] = bhi[j]; }
    }

    // --- sweep row tiles ---
    #pragma unroll
    for (int t = 0; t < MTILES_PER_WAVE; ++t) {
        const int row0 = (blockIdx.x * MTILES_PER_WAVE + t) << 4;
        if (row0 >= M) break;                  // uniform across the wave
        const __bf16* arow = A + (size_t)(row0 + m) * 128;
        v8f c = {};
        #pragma unroll
        for (int s = 0; s < 4; ++s) {
            const int kk = s * 32;
            v8bf alo = *(const v8bf*)(arow + kk + khalf * 8);
            v8bf ahi = *(const v8bf*)(arow + kk + 16 + khalf * 8);
            v16bf a;
            #pragma unroll
            for (int j = 0; j < 8; ++j) { a[j] = alo[j]; a[8 + j] = ahi[j]; }
            c = __builtin_amdgcn_wmma_f32_16x16x32_bf16(false, a, false, bfrag[s],
                                                        (short)0, c, false, false);
        }
        // D layout: VGPR r -> M = r + 8*khalf, N = lane&15
        #pragma unroll
        for (int r = 0; r < 8; ++r)
            C[(size_t)(row0 + khalf * 8 + r) * 128 + n0 + m] = c[r];
    }
}

// one wave per edge: agg[dst] += hw[src] * dis[src]*dis[dst]
__global__ void scatter_k(const float* __restrict__ hw,
                          const int* __restrict__ src, const int* __restrict__ dst,
                          const float* __restrict__ dis,
                          float* __restrict__ agg, int E) {
    int e = blockIdx.x * 8 + (threadIdx.x >> 5);
    if (e >= E) return;
    int lane = threadIdx.x & 31;
    int s = src[e], d = dst[e];
    float norm = dis[s] * dis[d];
    const float4 v = ((const float4*)(hw + (size_t)s * 128))[lane];
    float* ap = agg + (size_t)d * 128 + lane * 4;
    atomicAdd(ap + 0, v.x * norm);
    atomicAdd(ap + 1, v.y * norm);
    atomicAdd(ap + 2, v.z * norm);
    atomicAdd(ap + 3, v.w * norm);
}

// h_out (in agg, in-place) = [relu]( agg + hw * dis^2 + bias )
__global__ void combine_k(float* __restrict__ agg, const float* __restrict__ hw,
                          const float* __restrict__ dis, const float* __restrict__ bias,
                          int N, int do_relu) {
    size_t i = (size_t)blockIdx.x * blockDim.x + threadIdx.x;
    if (i >= (size_t)N * 128) return;
    int node = (int)(i >> 7), f = (int)(i & 127);
    float ds = dis[node];
    float v = agg[i] + hw[i] * ds * ds + bias[f];
    if (do_relu) v = fmaxf(v, 0.0f);
    agg[i] = v;
}

// one wave per node: sums[batch[n]] += h[n]; cnt[batch[n]] += 1
__global__ void pool_k(const float* __restrict__ h, const int* __restrict__ batch,
                       float* __restrict__ sums, float* __restrict__ cnt, int N) {
    int n = blockIdx.x * 8 + (threadIdx.x >> 5);
    if (n >= N) return;
    int lane = threadIdx.x & 31;
    int g = batch[n];
    const float4 v = ((const float4*)(h + (size_t)n * 128))[lane];
    float* sp = sums + (size_t)g * 128 + lane * 4;
    atomicAdd(sp + 0, v.x);
    atomicAdd(sp + 1, v.y);
    atomicAdd(sp + 2, v.z);
    atomicAdd(sp + 3, v.w);
    if (lane == 0) atomicAdd(&cnt[g], 1.0f);
}

// out[g,c] = (sums[g,:]/max(cnt,1)) @ linW[:,c] + linb[c]   (64x10, one block)
__global__ void final_k(const float* __restrict__ sums, const float* __restrict__ cnt,
                        const float* __restrict__ linW, const float* __restrict__ linb,
                        float* __restrict__ out) {
    int t = threadIdx.x;
    if (t >= N_GRAPHS * N_CLASSES) return;
    int g = t / N_CLASSES, c = t % N_CLASSES;
    float invc = 1.0f / fmaxf(cnt[g], 1.0f);
    float acc = linb[c];
    #pragma unroll 4
    for (int f = 0; f < 128; ++f)
        acc += sums[g * 128 + f] * invc * linW[f * N_CLASSES + c];
    out[g * N_CLASSES + c] = acc;
}

// ---------- launch ----------
extern "C" void kernel_launch(void* const* d_in, const int* in_sizes, int n_in,
                              void* d_out, int out_size, void* d_ws, size_t ws_size,
                              hipStream_t stream) {
    const float* x       = (const float*)d_in[0];
    const int*   eidx    = (const int*)d_in[1];       // [2, E] int32
    const int*   batch   = (const int*)d_in[2];
    const float* W1      = (const float*)d_in[3];
    const float* b1      = (const float*)d_in[4];
    const float* W2      = (const float*)d_in[5];
    const float* b2      = (const float*)d_in[6];
    const float* W3      = (const float*)d_in[7];
    const float* b3      = (const float*)d_in[8];
    const float* linW    = (const float*)d_in[9];
    const float* linb    = (const float*)d_in[10];
    float* out = (float*)d_out;

    const int* src = eidx;
    const int* dst = eidx + N_EDGES;

    // workspace layout (256B-aligned regions)
    const size_t SZ_H   = (size_t)N_NODES * 128 * sizeof(float);   // 51,200,000
    const size_t SZ_HB  = (size_t)N_NODES * 128 * sizeof(__bf16);  // 25,600,000
    const size_t SZ_DEG = ((size_t)N_NODES * sizeof(float) + 255) & ~(size_t)255;
    const size_t SZ_WT  = (size_t)3 * 128 * 128 * sizeof(__bf16);
    char* ws = (char*)d_ws;
    float*  P    = (float*)(ws);                                 // agg / layer output
    float*  Q    = (float*)(ws + SZ_H);                          // GEMM result (hw)
    __bf16* HB   = (__bf16*)(ws + 2 * SZ_H);                     // bf16 activations
    float*  deg  = (float*)(ws + 2 * SZ_H + SZ_HB);
    float*  dis  = (float*)(ws + 2 * SZ_H + SZ_HB + SZ_DEG);
    __bf16* WT   = (__bf16*)(ws + 2 * SZ_H + SZ_HB + 2 * SZ_DEG);
    float*  sums = (float*)(ws + 2 * SZ_H + SZ_HB + 2 * SZ_DEG + SZ_WT);
    float*  cnt  = sums + N_GRAPHS * 128;
    if (ws_size < 2 * SZ_H + SZ_HB + 2 * SZ_DEG + SZ_WT + (N_GRAPHS * 128 + N_GRAPHS) * 4) return;

    const size_t NH = (size_t)N_NODES * 128;
    const int BLK = 256;
    const int gNH   = (int)((NH + BLK - 1) / BLK);
    const int gN    = (N_NODES + BLK - 1) / BLK;
    const int gE    = (N_EDGES + BLK - 1) / BLK;
    const int gEdgeWave = (N_EDGES + 7) / 8;
    const int gNodeWave = (N_NODES + 7) / 8;
    const int nMTiles = N_NODES / 16;                             // 6250
    const int gGemm = (nMTiles + MTILES_PER_WAVE - 1) / MTILES_PER_WAVE;   // 1563

    // degree + normalization
    zero_f32<<<gN, BLK, 0, stream>>>(deg, N_NODES);
    degree_k<<<gE, BLK, 0, stream>>>(dst, deg, N_EDGES);
    dis_k<<<gN, BLK, 0, stream>>>(deg, dis, N_NODES);

    // weight prep (transpose + bf16)
    prep_weight_k<<<64, BLK, 0, stream>>>(W1, WT);
    prep_weight_k<<<64, BLK, 0, stream>>>(W2, WT + 128 * 128);
    prep_weight_k<<<64, BLK, 0, stream>>>(W3, WT + 2 * 128 * 128);

    const float* layer_in[3] = { x, P, P };
    const __bf16* layer_wt[3] = { WT, WT + 128 * 128, WT + 2 * 128 * 128 };
    const float* layer_b[3] = { b1, b2, b3 };

    for (int l = 0; l < 3; ++l) {
        f32_to_bf16_k<<<gNH, BLK, 0, stream>>>(layer_in[l], HB, NH);
        gemm_wmma_bf16<<<gGemm, BLK, 0, stream>>>(HB, layer_wt[l], Q, N_NODES);
        zero_f32<<<gNH, BLK, 0, stream>>>(P, NH);
        scatter_k<<<gEdgeWave, BLK, 0, stream>>>(Q, src, dst, dis, P, N_EDGES);
        combine_k<<<gNH, BLK, 0, stream>>>(P, Q, dis, layer_b[l], N_NODES, l < 2 ? 1 : 0);
    }

    // global mean pool + classifier
    zero_f32<<<(N_GRAPHS * 128 + N_GRAPHS + BLK - 1) / BLK, BLK, 0, stream>>>(sums, N_GRAPHS * 128 + N_GRAPHS);
    pool_k<<<gNodeWave, BLK, 0, stream>>>(P, batch, sums, cnt, N_NODES);
    final_k<<<1, 640, 0, stream>>>(sums, cnt, linW, linb, out);
}
